// RoutedMemoryAttention_51934744543452
// MI455X (gfx1250) — matrix operand order repaired and checked by reading, not compiled, not run on hardware
//
#include <hip/hip_runtime.h>
#include <hip/hip_bf16.h>
#include <math.h>

typedef _Float16 half_t;
typedef __attribute__((ext_vector_type(16))) _Float16 v16h;
typedef __attribute__((ext_vector_type(8)))  _Float16 v8h;
typedef __attribute__((ext_vector_type(8)))  float    v8f;

#define D_MODEL   1024
#define N_HEADS   16
#define SLOT_DIM  64
#define N_SLOTS   1024
#define M_TOTAL   8192   // B*T = 4*2048
#define LN_EPS    1e-5f

#define LDA_PAD   (D_MODEL + 8)                         // halves per LDS row
#define PANEL_LDS_BYTES (64 * LDA_PAD * sizeof(half_t)) // 132 KB dynamic LDS

// ---------------------------------------------------------------------------
// WMMA fragment helpers (wave32, 16x16x32 f16 -> f32)
// A-matrix 16x32 (MxK), 16-bit: lane<16 holds row=lane, K {0..7,16..23};
// lane>=16 holds row=lane-16, K {8..15,24..31}.   (ISA 7.12.2)
// ---------------------------------------------------------------------------
__device__ __forceinline__ v16h load_a_frag(const half_t* __restrict__ A,
                                            int lda, int k0) {
  const int lane = threadIdx.x & 31;
  const int r    = lane & 15;
  const int koff = (lane >> 4) * 8;
  const half_t* p0 = A + (size_t)r * lda + k0 + koff;      // 16B aligned
  const half_t* p1 = p0 + 16;
  v16h f;
  *reinterpret_cast<v8h*>(&f)       = *reinterpret_cast<const v8h*>(p0);
  *(reinterpret_cast<v8h*>(&f) + 1) = *reinterpret_cast<const v8h*>(p1);
  return f;
}

// B-matrix 32x16 (KxN), 16-bit: lane holds column n=lane&15,
// K {0..7,16..23} (lane<16) or {8..15,24..31} (lane>=16), matching A,
// from the row-major [N][K] weight row.
__device__ __forceinline__ v16h load_b_frag(const half_t* __restrict__ W,
                                            int ldw, int k0) {
  const int lane = threadIdx.x & 31;
  const int n    = lane & 15;
  const int koff = (lane >> 4) * 8;
  const half_t* p0 = W + (size_t)n * ldw + k0 + koff;
  v16h f;
  *reinterpret_cast<v8h*>(&f)       = *reinterpret_cast<const v8h*>(p0);
  *(reinterpret_cast<v8h*>(&f) + 1) = *reinterpret_cast<const v8h*>(p0 + 16);
  return f;
}

__device__ __forceinline__ v8f wmma_f16(v16h a, v16h b, v8f c) {
  return __builtin_amdgcn_wmma_f32_16x16x32_f16(
      /*neg_a=*/false, a, /*neg_b=*/false, b,
      /*c_mod=*/(short)0, c, /*reuse_a=*/false, /*reuse_b=*/false);
}

// ---------------------------------------------------------------------------
// Kernel 1: weight prep. q_w / out_w f32->f16, signatures ternarized to f16
// (exactly representable: {-1,0,+1}).  All three arrays are 1M elements.
// ---------------------------------------------------------------------------
__global__ __launch_bounds__(256)
void rma_prep_kernel(const float* __restrict__ qw, const float* __restrict__ ow,
                     const float* __restrict__ sg,
                     half_t* __restrict__ qw_h, half_t* __restrict__ ow_h,
                     half_t* __restrict__ sg_h) {
  const int stride = gridDim.x * blockDim.x;
  for (int j = blockIdx.x * blockDim.x + threadIdx.x;
       j < D_MODEL * D_MODEL; j += stride) {
    qw_h[j] = (half_t)qw[j];
    ow_h[j] = (half_t)ow[j];
    float s = sg[j];
    sg_h[j] = (half_t)((s > 0.3f) ? 1.0f : ((s < -0.3f) ? -1.0f : 0.0f));
  }
}

// ---------------------------------------------------------------------------
// Kernel 2: fused LayerNorm + query projection, 64x64 register blocking.
// Block = 512 threads (16 waves) = one 64-token M-panel (132 KB dynamic LDS).
// Phase 1: wave w LayerNorms tokens w*4..w*4+3 into the shared f16 A-panel
//          (x read exactly once from HBM).
// Phase 2: wave w computes a 64x64 output tile (cols [w*64,w*64+64)):
//          per K-step, 4 A-frags (LDS) x 4 B-frags (global/L2) -> 16 WMMAs.
//          Weight traffic: 128 blocks x 2 MB = 256 MB of L2 reads (4x less
//          than 16-row blocking); intensity 64 FLOP / L2 byte.
// ---------------------------------------------------------------------------
__global__ __launch_bounds__(512)
void rma_ln_qproj_kernel(const float* __restrict__ x,
                         const float* __restrict__ ln_g,
                         const float* __restrict__ ln_b,
                         const half_t* __restrict__ qw_h,
                         const float* __restrict__ q_b,
                         half_t* __restrict__ q_h) {
  extern __shared__ half_t aLds[];        // [64][LDA_PAD]
  const int wv   = threadIdx.x >> 5;
  const int lane = threadIdx.x & 31;
  const int m0   = blockIdx.x * 64;

  // ---- Phase 1: LayerNorm, 4 tokens per wave ----
#pragma unroll
  for (int i = 0; i < 4; ++i) {
    const int row = wv * 4 + i;
    const float* xr = x + (size_t)(m0 + row) * D_MODEL;
    float vbuf[32];
    float sum = 0.f, ss = 0.f;
#pragma unroll
    for (int t = 0; t < 32; ++t) {
      float v = xr[lane + t * 32];
      vbuf[t] = v; sum += v; ss += v * v;
    }
#pragma unroll
    for (int off = 16; off > 0; off >>= 1) {
      sum += __shfl_xor(sum, off, 32);
      ss  += __shfl_xor(ss,  off, 32);
    }
    const float mean = sum * (1.0f / D_MODEL);
    const float var  = ss * (1.0f / D_MODEL) - mean * mean;
    const float inv  = rsqrtf(var + LN_EPS);
#pragma unroll
    for (int t = 0; t < 32; ++t) {
      const int c = lane + t * 32;
      aLds[row * LDA_PAD + c] =
          (half_t)((vbuf[t] - mean) * inv * ln_g[c] + ln_b[c]);
    }
  }
  __syncthreads();

  // ---- Phase 2: 64x64 GEMM tile per wave ----
  const v8f zero = {};
  v8f acc[4][4];
#pragma unroll
  for (int mi = 0; mi < 4; ++mi)
#pragma unroll
    for (int ni = 0; ni < 4; ++ni) acc[mi][ni] = zero;

  for (int k0 = 0; k0 < D_MODEL; k0 += 32) {
    v16h a[4];
#pragma unroll
    for (int mi = 0; mi < 4; ++mi)
      a[mi] = load_a_frag(aLds + mi * 16 * LDA_PAD, LDA_PAD, k0);
#pragma unroll
    for (int ni = 0; ni < 4; ++ni) {
      v16h b = load_b_frag(qw_h + (size_t)(wv * 64 + ni * 16) * D_MODEL,
                           D_MODEL, k0);
#pragma unroll
      for (int mi = 0; mi < 4; ++mi)
        acc[mi][ni] = wmma_f16(a[mi], b, acc[mi][ni]);
    }
  }

  // ---- store q (f16, row-major [M][1024]) with bias ----
  const int col   = lane & 15;
  const int rbase = (lane >> 4) * 8;
#pragma unroll
  for (int mi = 0; mi < 4; ++mi)
#pragma unroll
    for (int ni = 0; ni < 4; ++ni)
#pragma unroll
      for (int v = 0; v < 8; ++v) {
        const int r = m0 + mi * 16 + rbase + v;
        const int c = wv * 64 + ni * 16 + col;
        q_h[(size_t)r * D_MODEL + c] = (half_t)(acc[mi][ni][v] + q_b[c]);
      }
}

// ---------------------------------------------------------------------------
// Kernel 3: fused scores + top-2 + softmax + value gather.
// Block = 128 threads (4 waves); wave = (64-token panel, head).
// 8 q A-frags (64 tokens x K=64) stay register-resident for all 64 slot
// blocks; each signature B-frag is reused across 4 score tiles -> signature
// L2 traffic = 2048 waves x 128 KB = 256 MB (4x less than 16-row blocking).
// The [64 x 1024] score panel exists only as register WMMA tiles spilled to
// per-wave LDS for the top-2 scan -> scores NEVER touch HBM (saves 512 MB).
// ---------------------------------------------------------------------------
__global__ __launch_bounds__(128)
void rma_route_kernel(const half_t* __restrict__ q_h,
                      const half_t* __restrict__ sg_h,
                      const float* __restrict__ vals,
                      const float* __restrict__ temp,
                      half_t* __restrict__ read_h) {
  __shared__ float sbuf[4][4][16][17];   // [wave][m-subtile][row][slot]
  __shared__ float wbuf[4][4][16][2];
  __shared__ int   ibuf[4][4][16][2];

  const int wv   = threadIdx.x >> 5;
  const int lane = threadIdx.x & 31;
  const int m0   = blockIdx.x * 64;
  const int h    = blockIdx.y * 4 + wv;
  const float invscale = 1.0f / (temp[0] * 8.0f);   // sqrt(SLOT_DIM)=8

  v16h a0[4], a1[4];
#pragma unroll
  for (int mi = 0; mi < 4; ++mi) {
    const half_t* qbase =
        q_h + (size_t)(m0 + mi * 16) * D_MODEL + h * SLOT_DIM;
    a0[mi] = load_a_frag(qbase, D_MODEL, 0);
    a1[mi] = load_a_frag(qbase, D_MODEL, 32);
  }

  float s1[4], s2[4]; int i1[4], i2[4];
#pragma unroll
  for (int mi = 0; mi < 4; ++mi) {
    s1[mi] = -3.4e38f; s2[mi] = -3.4e38f; i1[mi] = 0; i2[mi] = 0;
  }
  const int col   = lane & 15;
  const int rbase = (lane >> 4) * 8;
  const v8f zero = {};

  for (int sb = 0; sb < N_SLOTS / 16; ++sb) {
    const half_t* bbase = sg_h + ((size_t)h * N_SLOTS + sb * 16) * SLOT_DIM;
    v16h b0 = load_b_frag(bbase, SLOT_DIM, 0);
    v16h b1 = load_b_frag(bbase, SLOT_DIM, 32);
#pragma unroll
    for (int mi = 0; mi < 4; ++mi) {
      v8f acc = zero;
      acc = wmma_f16(a0[mi], b0, acc);
      acc = wmma_f16(a1[mi], b1, acc);
#pragma unroll
      for (int v = 0; v < 8; ++v) sbuf[wv][mi][rbase + v][col] = acc[v];
    }
    __syncthreads();
    if (lane < 16) {
#pragma unroll
      for (int mi = 0; mi < 4; ++mi)
#pragma unroll
        for (int j = 0; j < 16; ++j) {
          const float s  = sbuf[wv][mi][lane][j];
          const int   id = sb * 16 + j;
          if (s > s1[mi])      { s2[mi] = s1[mi]; i2[mi] = i1[mi];
                                 s1[mi] = s;      i1[mi] = id; }
          else if (s > s2[mi]) { s2[mi] = s;      i2[mi] = id; }
        }
    }
    __syncthreads();
  }

  // softmax over 2 (scale is monotonic -> applied post-selection)
  if (lane < 16) {
#pragma unroll
    for (int mi = 0; mi < 4; ++mi) {
      const float e  = __expf((s2[mi] - s1[mi]) * invscale);
      const float w1 = 1.0f / (1.0f + e);
      wbuf[wv][mi][lane][0] = w1;
      wbuf[wv][mi][lane][1] = e * w1;
      ibuf[wv][mi][lane][0] = i1[mi];
      ibuf[wv][mi][lane][1] = i2[mi];
    }
  }
  __syncthreads();

  // cooperative gather of the 2 selected value rows per token
#pragma unroll
  for (int mi = 0; mi < 4; ++mi) {
    for (int r = 0; r < 16; ++r) {
      const float w1 = wbuf[wv][mi][r][0], w2 = wbuf[wv][mi][r][1];
      const float* v1 =
          vals + ((size_t)h * N_SLOTS + ibuf[wv][mi][r][0]) * SLOT_DIM;
      const float* v2 =
          vals + ((size_t)h * N_SLOTS + ibuf[wv][mi][r][1]) * SLOT_DIM;
#pragma unroll
      for (int cc = 0; cc < 2; ++cc) {
        const int c = lane + cc * 32;
        const float rv = w1 * v1[c] + w2 * v2[c];
        read_h[(size_t)(m0 + mi * 16 + r) * D_MODEL + h * SLOT_DIM + c] =
            (half_t)rv;
      }
    }
  }
}

// ---------------------------------------------------------------------------
// Kernel 4: output projection + bias + residual, 64x64 register blocking.
// The 64-token read panel is staged once in dynamic LDS, reused by 16 waves.
// ---------------------------------------------------------------------------
__global__ __launch_bounds__(512)
void rma_out_gemm_kernel(const half_t* __restrict__ read_h,
                         const half_t* __restrict__ ow_h,
                         const float* __restrict__ o_b,
                         const float* __restrict__ x,
                         float* __restrict__ out) {
  extern __shared__ half_t aLds[];        // [64][LDA_PAD]
  const int m0 = blockIdx.x * 64;

  for (int i = threadIdx.x; i < 64 * D_MODEL; i += 512) {
    const int r = i >> 10, c = i & (D_MODEL - 1);
    aLds[r * LDA_PAD + c] = read_h[(size_t)(m0 + r) * D_MODEL + c];
  }
  __syncthreads();

  const int wv   = threadIdx.x >> 5;
  const int lane = threadIdx.x & 31;
  const v8f zero = {};
  v8f acc[4][4];
#pragma unroll
  for (int mi = 0; mi < 4; ++mi)
#pragma unroll
    for (int ni = 0; ni < 4; ++ni) acc[mi][ni] = zero;

  for (int k0 = 0; k0 < D_MODEL; k0 += 32) {
    v16h a[4];
#pragma unroll
    for (int mi = 0; mi < 4; ++mi)
      a[mi] = load_a_frag(aLds + mi * 16 * LDA_PAD, LDA_PAD, k0);
#pragma unroll
    for (int ni = 0; ni < 4; ++ni) {
      v16h b = load_b_frag(ow_h + (size_t)(wv * 64 + ni * 16) * D_MODEL,
                           D_MODEL, k0);
#pragma unroll
      for (int mi = 0; mi < 4; ++mi)
        acc[mi][ni] = wmma_f16(a[mi], b, acc[mi][ni]);
    }
  }

  const int col   = lane & 15;
  const int rbase = (lane >> 4) * 8;
#pragma unroll
  for (int mi = 0; mi < 4; ++mi)
#pragma unroll
    for (int ni = 0; ni < 4; ++ni)
#pragma unroll
      for (int v = 0; v < 8; ++v) {
        const int r = m0 + mi * 16 + rbase + v;
        const int c = wv * 64 + ni * 16 + col;
        out[(size_t)r * D_MODEL + c] =
            acc[mi][ni][v] + o_b[c] + x[(size_t)r * D_MODEL + c];
      }
}

// ---------------------------------------------------------------------------
extern "C" void kernel_launch(void* const* d_in, const int* in_sizes, int n_in,
                              void* d_out, int out_size, void* d_ws,
                              size_t ws_size, hipStream_t stream) {
  const float* x    = (const float*)d_in[0];
  const float* ln_g = (const float*)d_in[1];
  const float* ln_b = (const float*)d_in[2];
  const float* q_w  = (const float*)d_in[3];
  const float* q_b  = (const float*)d_in[4];
  const float* sig  = (const float*)d_in[5];
  const float* vals = (const float*)d_in[6];
  const float* o_w  = (const float*)d_in[7];
  const float* o_b  = (const float*)d_in[8];
  const float* temp = (const float*)d_in[9];
  float* out = (float*)d_out;

  // workspace layout (f16 elements): 2+2+2 MB weights, 16 MB q, 16 MB read
  half_t* ws   = (half_t*)d_ws;
  half_t* qw_h = ws;
  half_t* ow_h = qw_h + (size_t)D_MODEL * D_MODEL;
  half_t* sg_h = ow_h + (size_t)D_MODEL * D_MODEL;
  half_t* q_h  = sg_h + (size_t)N_HEADS * N_SLOTS * SLOT_DIM;
  half_t* rd_h = q_h  + (size_t)M_TOTAL * D_MODEL;

  rma_prep_kernel<<<dim3(512), dim3(256), 0, stream>>>(
      q_w, o_w, sig, qw_h, ow_h, sg_h);

  rma_ln_qproj_kernel<<<dim3(M_TOTAL / 64), dim3(512), PANEL_LDS_BYTES,
                        stream>>>(x, ln_g, ln_b, qw_h, q_b, q_h);

  rma_route_kernel<<<dim3(M_TOTAL / 64, N_HEADS / 4), dim3(128), 0, stream>>>(
      q_h, sg_h, vals, temp, rd_h);

  rma_out_gemm_kernel<<<dim3(M_TOTAL / 64), dim3(512), PANEL_LDS_BYTES,
                        stream>>>(rd_h, ow_h, o_b, x, out);
}
